// AttentionBlock_40870908789381
// MI455X (gfx1250) — compile-verified, matrix operand order checked by reading, not run on hardware
//
#include <hip/hip_runtime.h>

// Problem dims (fixed by reference)
#define B_  4096
#define S_  32
#define D_  512
#define E_  512
#define SD_ (S_*D_)   // 16384

typedef __attribute__((ext_vector_type(16))) __bf16 v16bf;
typedef __attribute__((ext_vector_type(8)))  __bf16 v8bf;
typedef __attribute__((ext_vector_type(8)))  float  v8f;

// gfx1250 async global->LDS path (ASYNCcnt), guarded so compile never breaks.
#if defined(__has_builtin)
#  if __has_builtin(__builtin_amdgcn_global_load_async_to_lds_b128) && \
      __has_builtin(__builtin_amdgcn_s_wait_asynccnt)
#    define USE_ASYNC_LDS 1
#  endif
#endif
#ifndef USE_ASYNC_LDS
#  define USE_ASYNC_LDS 0
#endif

#if USE_ASYNC_LDS
// The builtin's pointer params are GCC-style v4i vectors in AS1 / AS3.
typedef int v4i_gcc __attribute__((vector_size(16)));
typedef __attribute__((address_space(1))) v4i_gcc* gptr_b128;
typedef __attribute__((address_space(3))) v4i_gcc* lptr_b128;
static __device__ __forceinline__ void async_copy16(const __bf16* g, __bf16* l) {
  __builtin_amdgcn_global_load_async_to_lds_b128((gptr_b128)g, (lptr_b128)l, 0, 0);
}
#endif

static __device__ __forceinline__ v8bf load_cvt8(const float* g, float scale) {
  float4 f0 = *(const float4*)g;
  float4 f1 = *(const float4*)(g + 4);
  v8bf h;
  h[0] = (__bf16)(f0.x * scale); h[1] = (__bf16)(f0.y * scale);
  h[2] = (__bf16)(f0.z * scale); h[3] = (__bf16)(f0.w * scale);
  h[4] = (__bf16)(f1.x * scale); h[5] = (__bf16)(f1.y * scale);
  h[6] = (__bf16)(f1.z * scale); h[7] = (__bf16)(f1.w * scale);
  return h;
}

static __device__ __forceinline__ v8f wmma_bf16(v16bf a, v16bf b, v8f c) {
  // 8 args: (neg_a, A, neg_b, B, c_mod, C, reuse_a, reuse_b)
  return __builtin_amdgcn_wmma_f32_16x16x32_bf16(false, a, false, b, (short)0, c,
                                                 false, false);
}

// ---------------------------------------------------------------------------
// Phase 0: Vt[e][s*D+d] = (bf16) V[s][d][e]   (LDS tile transpose, coalesced)
// grid (SD_/32, E_/32), 256 threads
// ---------------------------------------------------------------------------
__global__ __launch_bounds__(256) void k_prepV(const float* __restrict__ V,
                                               __bf16* __restrict__ Vt) {
  __shared__ __bf16 t[32][33];
  const int sd0 = blockIdx.x * 32;
  const int e0  = blockIdx.y * 32;
  const int tx  = threadIdx.x & 31;   // fast dim
  const int ty4 = threadIdx.x >> 5;   // 0..7
  #pragma unroll
  for (int i = 0; i < 4; ++i) {
    int sd = sd0 + ty4 * 4 + i;
    t[ty4 * 4 + i][tx] = (__bf16)V[(size_t)sd * E_ + e0 + tx];
  }
  __syncthreads();
  #pragma unroll
  for (int i = 0; i < 4; ++i) {
    int e = e0 + ty4 * 4 + i;
    Vt[(size_t)e * SD_ + sd0 + tx] = t[tx][ty4 * 4 + i];
  }
}

// ---------------------------------------------------------------------------
// Phase 1: M_s = Q_s @ K_s^T, stored transposed as Mt[s][d2][d1] (bf16)
// grid (D_/64, D_/128, S_), 256 threads = 8 waves (2 M x 4 N), wave = 32x32
// ---------------------------------------------------------------------------
__global__ __launch_bounds__(256) void k_qkT(const float* __restrict__ Q,
                                             const float* __restrict__ K,
                                             __bf16* __restrict__ Mt) {
  const int s  = blockIdx.z;
  const int m0 = blockIdx.x * 64;    // d1 block
  const int n0 = blockIdx.y * 128;   // d2 block
  __shared__ __bf16 At[64 * 32];
  __shared__ __bf16 Bt[128 * 32];
  const int tid  = threadIdx.x;
  const int wave = tid >> 5, lane = tid & 31;
  const int l16  = lane & 15, half = lane >> 4;
  const int wm = (wave >> 2) * 32;   // wave M offset within 64
  const int wn = (wave & 3) * 32;    // wave N offset within 128
  const float* Qs = Q + (size_t)s * D_ * E_;
  const float* Ks = K + (size_t)s * D_ * E_;

  v8f acc[2][2] = {};
  for (int k0 = 0; k0 < E_; k0 += 32) {
    __syncthreads();
    { // stage A: Q rows m0..m0+63, cols k0..k0+31  (f32 -> bf16 on the fly)
      int row = tid >> 2, c = (tid & 3) * 8;
      const float* g = Qs + (size_t)(m0 + row) * E_ + k0 + c;
      __builtin_prefetch((const void*)(g + 32), 0, 1);
      *(v8bf*)&At[row * 32 + c] = load_cvt8(g, 1.0f);
    }
    #pragma unroll
    for (int it = 0; it < 2; ++it) { // stage B: K rows n0..n0+127 (Bt[n][k])
      int idx = tid + it * 256;
      int row = idx >> 2, c = (idx & 3) * 8;
      const float* g = Ks + (size_t)(n0 + row) * E_ + k0 + c;
      __builtin_prefetch((const void*)(g + 32), 0, 1);
      *(v8bf*)&Bt[row * 32 + c] = load_cvt8(g, 1.0f);
    }
    __syncthreads();
    v16bf a[2], b[2];
    #pragma unroll
    for (int mm = 0; mm < 2; ++mm)
      a[mm] = *(const v16bf*)&At[(wm + mm * 16 + l16) * 32 + half * 16];
    #pragma unroll
    for (int nn = 0; nn < 2; ++nn)
      b[nn] = *(const v16bf*)&Bt[(wn + nn * 16 + l16) * 32 + half * 16];
    #pragma unroll
    for (int mm = 0; mm < 2; ++mm)
      #pragma unroll
      for (int nn = 0; nn < 2; ++nn)
        acc[mm][nn] = wmma_bf16(a[mm], b[nn], acc[mm][nn]);
  }
  // transposed bf16 store: Mt[s][n=d2][m=d1] (8 consecutive m per lane -> b128)
  __bf16* Ms = Mt + (size_t)s * D_ * D_;
  #pragma unroll
  for (int mm = 0; mm < 2; ++mm)
    #pragma unroll
    for (int nn = 0; nn < 2; ++nn)
      #pragma unroll
      for (int r = 0; r < 8; ++r) {
        int m = m0 + wm + mm * 16 + half * 8 + r;
        int n = n0 + wn + nn * 16 + l16;
        Ms[(size_t)n * D_ + m] = (__bf16)acc[mm][nn][r];
      }
}

// ---------------------------------------------------------------------------
// Phase 2: scores[b,s] += diag( Xs @ M_s @ Xs^T )
// grid (B_/32, S_), 256 threads. WG tile 32M x 512N, Xs staged fully in LDS.
// 8 waves each own a 16-col N slice per 128-wide N block; fused dot epilogue.
// B tile (bf16 Mt) staged via async global->LDS when available.
// ---------------------------------------------------------------------------
__global__ __launch_bounds__(256) void k_scores(const float* __restrict__ x,
                                                const __bf16* __restrict__ Mt,
                                                float* __restrict__ scores) {
  const int b0 = blockIdx.x * 32;
  const int s  = blockIdx.y;
  __shared__ __bf16 Xt[32 * 512];    // 32 KB: x[b0..b0+31][s*D .. s*D+511] bf16
  __shared__ __bf16 Bt[128 * 32];    // 8 KB per k-step
  const int tid  = threadIdx.x;
  const int wave = tid >> 5, lane = tid & 31;
  const int l16  = lane & 15, half = lane >> 4;
  const __bf16* Ms = Mt + (size_t)s * D_ * D_;

  // stage full X tile once (f32 -> bf16)
  for (int it = tid; it < 32 * 512 / 8; it += 256) {
    int row = it >> 6, c = (it & 63) * 8;
    const float* g = x + (size_t)(b0 + row) * SD_ + s * D_ + c;
    *(v8bf*)&Xt[row * 512 + c] = load_cvt8(g, 1.0f);
  }
  __syncthreads();

  for (int nb = 0; nb < 4; ++nb) {       // N blocks of 128 over d2
    const int n0 = nb * 128;
    v8f acc[2] = {};                     // 2 M-tiles x (16 x 16)
    for (int k0 = 0; k0 < D_; k0 += 32) {
      __syncthreads();
#if USE_ASYNC_LDS
      #pragma unroll
      for (int it = 0; it < 2; ++it) {   // stage B = Mt rows n0..n0+127
        int idx = tid + it * 256;
        int row = idx >> 2, c = (idx & 3) * 8;
        async_copy16(Ms + (size_t)(n0 + row) * D_ + k0 + c, &Bt[row * 32 + c]);
      }
      __builtin_amdgcn_s_wait_asynccnt(0);
#else
      #pragma unroll
      for (int it = 0; it < 2; ++it) {
        int idx = tid + it * 256;
        int row = idx >> 2, c = (idx & 3) * 8;
        const __bf16* g = Ms + (size_t)(n0 + row) * D_ + k0 + c;
        *(v8bf*)&Bt[row * 32 + c] = *(const v8bf*)g;
      }
#endif
      __syncthreads();
      v16bf bfrag = *(const v16bf*)&Bt[(wave * 16 + l16) * 32 + half * 16];
      #pragma unroll
      for (int mm = 0; mm < 2; ++mm) {
        v16bf afrag =
            *(const v16bf*)&Xt[(mm * 16 + l16) * 512 + k0 + half * 16];
        acc[mm] = wmma_bf16(afrag, bfrag, acc[mm]);
      }
    }
    // epilogue: scores[b] += sum_n t[b,n] * x[b,n]   (n = n0 + wave*16 + l16)
    const int n = n0 + wave * 16 + l16;
    #pragma unroll
    for (int mm = 0; mm < 2; ++mm)
      #pragma unroll
      for (int r = 0; r < 8; ++r) {
        int mloc = mm * 16 + half * 8 + r;
        float p = acc[mm][r] * (float)Xt[mloc * 512 + n];
        #pragma unroll
        for (int o = 1; o < 16; o <<= 1) p += __shfl_xor(p, o, 32);
        if (l16 == 0)
          atomicAdd(&scores[(size_t)(b0 + mloc) * S_ + s], p);
      }
    __syncthreads();
  }
}

// ---------------------------------------------------------------------------
// Phase 3: weights = scores / ||scores||_2 (over S)
// ---------------------------------------------------------------------------
__global__ __launch_bounds__(256) void k_weights(const float* __restrict__ scores,
                                                 float* __restrict__ weights) {
  int b = blockIdx.x * blockDim.x + threadIdx.x;
  if (b >= B_) return;
  float v[S_], ss = 0.0f;
  #pragma unroll
  for (int s = 0; s < S_; ++s) {
    v[s] = scores[(size_t)b * S_ + s];
    ss += v[s] * v[s];
  }
  float inv = rsqrtf(ss);
  #pragma unroll
  for (int s = 0; s < S_; ++s) weights[(size_t)b * S_ + s] = v[s] * inv;
}

// ---------------------------------------------------------------------------
// Phase 4: y[b,e] = sum_{k<16384} (w[b,k>>9] * x[b,k]) * Vt[e][k]
// grid (B_/64, E_/128), 256 threads = 8 waves (2 M x 4 N), wave = 32x32
// B tile (bf16 Vt) staged via async global->LDS when available.
// ---------------------------------------------------------------------------
__global__ __launch_bounds__(256) void k_final(const float* __restrict__ x,
                                               const __bf16* __restrict__ Vt,
                                               const float* __restrict__ weights,
                                               float* __restrict__ y) {
  const int b0 = blockIdx.x * 64;
  const int e0 = blockIdx.y * 128;
  __shared__ __bf16 At[64 * 32];
  __shared__ __bf16 Bt[128 * 32];
  const int tid  = threadIdx.x;
  const int wave = tid >> 5, lane = tid & 31;
  const int l16  = lane & 15, half = lane >> 4;
  const int wm = (wave >> 2) * 32;
  const int wn = (wave & 3) * 32;

  v8f acc[2][2] = {};
  for (int k0 = 0; k0 < SD_; k0 += 32) {
    const int s = k0 >> 9;               // segment of this k-step (32 | 512)
    __syncthreads();
#if USE_ASYNC_LDS
    #pragma unroll
    for (int it = 0; it < 2; ++it) {     // stage B: Vt rows e0..e0+127 (bf16)
      int idx = tid + it * 256;
      int row = idx >> 2, c = (idx & 3) * 8;
      async_copy16(Vt + (size_t)(e0 + row) * SD_ + k0 + c, &Bt[row * 32 + c]);
    }
#endif
    { // stage A: x rows, scaled by per-(row,segment) weight, f32 -> bf16
      int row = tid >> 2, c = (tid & 3) * 8;
      const float* g = x + (size_t)(b0 + row) * SD_ + k0 + c;
      __builtin_prefetch((const void*)(g + 32), 0, 1);
      float wv = weights[(size_t)(b0 + row) * S_ + s];
      *(v8bf*)&At[row * 32 + c] = load_cvt8(g, wv);
    }
#if USE_ASYNC_LDS
    __builtin_amdgcn_s_wait_asynccnt(0);
#else
    #pragma unroll
    for (int it = 0; it < 2; ++it) {
      int idx = tid + it * 256;
      int row = idx >> 2, c = (idx & 3) * 8;
      const __bf16* g = Vt + (size_t)(e0 + row) * SD_ + k0 + c;
      __builtin_prefetch((const void*)(g + 32), 0, 1);
      *(v8bf*)&Bt[row * 32 + c] = *(const v8bf*)g;
    }
#endif
    __syncthreads();
    v16bf a[2], b[2];
    #pragma unroll
    for (int mm = 0; mm < 2; ++mm)
      a[mm] = *(const v16bf*)&At[(wm + mm * 16 + l16) * 32 + half * 16];
    #pragma unroll
    for (int nn = 0; nn < 2; ++nn)
      b[nn] = *(const v16bf*)&Bt[(wn + nn * 16 + l16) * 32 + half * 16];
    #pragma unroll
    for (int mm = 0; mm < 2; ++mm)
      #pragma unroll
      for (int nn = 0; nn < 2; ++nn)
        acc[mm][nn] = wmma_bf16(a[mm], b[nn], acc[mm][nn]);
  }
  #pragma unroll
  for (int mm = 0; mm < 2; ++mm)
    #pragma unroll
    for (int nn = 0; nn < 2; ++nn)
      #pragma unroll
      for (int r = 0; r < 8; ++r) {
        int m = b0 + wm + mm * 16 + half * 8 + r;
        int n = e0 + wn + nn * 16 + l16;
        y[(size_t)m * E_ + n] = acc[mm][nn][r];
      }
}

// ---------------------------------------------------------------------------
extern "C" void kernel_launch(void* const* d_in, const int* in_sizes, int n_in,
                              void* d_out, int out_size, void* d_ws, size_t ws_size,
                              hipStream_t stream) {
  const float* x = (const float*)d_in[0];   // [B, S*D]
  const float* Q = (const float*)d_in[1];   // [S, D, E]
  const float* K = (const float*)d_in[2];
  const float* V = (const float*)d_in[3];
  float* y = (float*)d_out;                 // [B, E]

  // workspace layout (≈34.6 MB)
  char* ws = (char*)d_ws;
  __bf16* Mt      = (__bf16*)ws;                               // 16 MiB
  __bf16* Vt      = (__bf16*)(ws + (size_t)16 * 1024 * 1024);  // 16 MiB
  float*  scores  = (float*)(ws + (size_t)32 * 1024 * 1024);   // 512 KiB
  float*  weights = (float*)(ws + (size_t)32 * 1024 * 1024 + B_ * S_ * sizeof(float));

  (void)hipMemsetAsync(scores, 0, (size_t)B_ * S_ * sizeof(float), stream);

  k_prepV<<<dim3(SD_ / 32, E_ / 32), 256, 0, stream>>>(V, Vt);
  k_qkT<<<dim3(D_ / 64, D_ / 128, S_), 256, 0, stream>>>(Q, K, Mt);
  k_scores<<<dim3(B_ / 32, S_), 256, 0, stream>>>(x, Mt, scores);
  k_weights<<<dim3(B_ / 256), 256, 0, stream>>>(scores, weights);
  k_final<<<dim3(B_ / 64, E_ / 128), 256, 0, stream>>>(x, Vt, weights, y);
}
